// BARTAttention_34583076667579
// MI455X (gfx1250) — matrix operand order repaired and checked by reading, not compiled, not run on hardware
//
#include <hip/hip_runtime.h>

#define D_MODEL 1024
#define N_HEADS 16
#define D_K     64
#define SEQ     2048
#define BATCH   4
#define M_TOTAL (BATCH * SEQ)   /* 8192 */

typedef __bf16 bf16;
typedef __attribute__((ext_vector_type(16))) __bf16 v16bf;
typedef __attribute__((ext_vector_type(8)))  float  v8f;

// GCC-style int4 vector (matches the async-builtin's declared pointee type)
typedef int v4i_gcc __attribute__((vector_size(16)));
typedef __attribute__((address_space(1))) v4i_gcc gas_v4i;   // global (__device__)
typedef __attribute__((address_space(3))) v4i_gcc las_v4i;   // LDS (__shared__)

union ABReg {
    unsigned int u[8];
    uint4 q128[2];
    v16bf v;
};

// ---------------------------------------------------------------------------
// Async global->LDS copy of one 16-byte chunk (gfx1250 ASYNCcnt path), with a
// synchronous fallback so the kernel compiles on any toolchain.
// ---------------------------------------------------------------------------
__device__ __forceinline__ void async_copy16(const bf16* g, bf16* l) {
#if __has_builtin(__builtin_amdgcn_global_load_async_to_lds_b128)
    __builtin_amdgcn_global_load_async_to_lds_b128(
        (gas_v4i*)(unsigned long long)(uintptr_t)g,
        (las_v4i*)(unsigned int)(uintptr_t)l, 0, 0);
#else
    *(uint4*)l = *(const uint4*)g;
#endif
}

__device__ __forceinline__ void wait_async() {
#if __has_builtin(__builtin_amdgcn_s_wait_asynccnt)
    __builtin_amdgcn_s_wait_asynccnt(0);
#elif __has_builtin(__builtin_amdgcn_global_load_async_to_lds_b128)
    asm volatile("s_wait_asynccnt 0" ::: "memory");
#endif
}

// ---------------------------------------------------------------------------
// WMMA helpers (wave32 layouts used by this code; A/B k order is contiguous)
// A tile 16x32 bf16: lane l -> row (l&15); elems 0..15 = K (l>>4)*16 .. +16
//                    (two b128 chunks)
// B tile 32x16 bf16 from K-major (transposed) storage: lane l -> col (l&15),
//                    16 contiguous K values starting at (l>>4)*16 (two b128)
// C tile 16x16 f32:  vgpr j -> row j + 8*(l>>4), col (l&15)
// ---------------------------------------------------------------------------
__device__ __forceinline__ v16bf load_tileA(const bf16* __restrict__ base, int stride) {
    const int l = threadIdx.x & 31;
    const bf16* p = base + (size_t)(l & 15) * stride + ((l >> 4) << 4);
    ABReg t;
    t.q128[0] = *(const uint4*)(p);
    t.q128[1] = *(const uint4*)(p + 8);
    return t.v;
}

__device__ __forceinline__ v16bf load_tileBT(const bf16* __restrict__ baseT, int stride) {
    const int l = threadIdx.x & 31;
    const bf16* p = baseT + (size_t)(l & 15) * stride + ((l >> 4) << 4);
    ABReg t;
    t.q128[0] = *(const uint4*)(p);
    t.q128[1] = *(const uint4*)(p + 8);
    return t.v;
}

// Same B-operand gather but from an LDS panel row-stride 40 bf16 (padded).
__device__ __forceinline__ v16bf load_tileB_lds(const bf16* base) {
    const int l = threadIdx.x & 31;
    const bf16* p = base + (l & 15) * 40 + ((l >> 4) << 4);
    ABReg t;
    t.q128[0] = *(const uint4*)(p);
    t.q128[1] = *(const uint4*)(p + 8);
    return t.v;
}

__device__ __forceinline__ v8f wmma_bf16(v16bf a, v16bf b, v8f c) {
    return __builtin_amdgcn_wmma_f32_16x16x32_bf16(false, a, false, b,
                                                   (short)0, c, false, false);
}

// ---------------------------------------------------------------------------
// Conversion kernels: f32 -> bf16 (x), f32 -> transposed bf16 (weights)
// ---------------------------------------------------------------------------
__global__ __launch_bounds__(256) void cvt_x_kernel(const float* __restrict__ src,
                                                    bf16* __restrict__ dst) {
    size_t i = (size_t)blockIdx.x * 256 + threadIdx.x;
    dst[i] = (bf16)src[i];
}

__global__ __launch_bounds__(256) void cvt_wT_kernel(const float* __restrict__ w,
                                                     bf16* __restrict__ wt) {
    size_t i = (size_t)blockIdx.x * 256 + threadIdx.x;   // i indexes wt[n*1024 + k]
    int n = (int)(i >> 10), k = (int)(i & 1023);
    wt[i] = (bf16)w[(size_t)k * D_MODEL + n];
}

// ---------------------------------------------------------------------------
// Projection GEMM: C[8192,1024] = A(bf16) @ W + bias. 256 threads = 8 waves,
// each wave owns a 32x64 C tile; the block's shared 32x64 B panel is staged
// in LDS with double-buffered async global->LDS copies (8x B reuse).
// MODE 0: Q -> [B,H,S,Dk] bf16, scaled 1/8    MODE 1: K -> [B,H,S,Dk] bf16
// MODE 2: V -> [B,H,Dk,S] bf16 (transposed)   MODE 3: out-proj -> f32 [M,1024]
// ---------------------------------------------------------------------------
template <int MODE>
__global__ __launch_bounds__(256) void gemm_kernel(const bf16* __restrict__ amat,
                                                   const bf16* __restrict__ wt,
                                                   const float* __restrict__ bias,
                                                   void* __restrict__ dstv) {
    __shared__ __align__(16) bf16 bpanel[2][64][40];   // 2 x 64 rows x 32k (+pad)

    const int wave = threadIdx.x >> 5;
    const int m0 = blockIdx.x * 256 + wave * 32;
    const int n0 = blockIdx.y * 64;

    // async B-panel producer: 256 threads x 16B = 4KB panel
    const int prow = threadIdx.x >> 2;          // 0..63 (column n within panel)
    const int pchk = (threadIdx.x & 3) << 3;    // k-chunk of 8 bf16
    const bf16* pg = wt + (size_t)(n0 + prow) * D_MODEL + pchk;

    v8f acc[2][4];
#pragma unroll
    for (int mi = 0; mi < 2; mi++)
#pragma unroll
        for (int ni = 0; ni < 4; ni++)
#pragma unroll
            for (int e = 0; e < 8; e++) acc[mi][ni][e] = 0.f;

    const bf16* aptr0 = amat + (size_t)m0 * D_MODEL;
    const bf16* aptr1 = amat + (size_t)(m0 + 16) * D_MODEL;

    // prologue: stage panel k=0 into buffer 0
    async_copy16(pg, &bpanel[0][prow][pchk]);
    wait_async();
    __syncthreads();

#pragma unroll 4
    for (int i = 0; i < D_MODEL / 32; i++) {
        const int kk = i * 32;
        const int cur = i & 1;
        if (kk + 32 < D_MODEL)   // uniform branch: prefetch next panel
            async_copy16(pg + kk + 32, &bpanel[cur ^ 1][prow][pchk]);

        __builtin_prefetch(aptr0 + kk + 128, 0, 0);
        __builtin_prefetch(aptr1 + kk + 128, 0, 0);
        v16bf a0 = load_tileA(aptr0 + kk, D_MODEL);
        v16bf a1 = load_tileA(aptr1 + kk, D_MODEL);
#pragma unroll
        for (int ni = 0; ni < 4; ni++) {
            v16bf bt = load_tileB_lds(&bpanel[cur][ni * 16][0]);
            acc[0][ni] = wmma_bf16(a0, bt, acc[0][ni]);
            acc[1][ni] = wmma_bf16(a1, bt, acc[1][ni]);
        }
        wait_async();     // own wave's async stores to LDS done
        __syncthreads();  // whole panel visible; safe to reuse buffers
    }

    const int l = threadIdx.x & 31;
    const int cn = l & 15, rhi = (l >> 4) << 3;
#pragma unroll
    for (int mi = 0; mi < 2; mi++) {
#pragma unroll
        for (int ni = 0; ni < 4; ni++) {
            const int n = n0 + ni * 16 + cn;
            const int h = n >> 6, d = n & 63;
            const float bval = bias[n];
#pragma unroll
            for (int j = 0; j < 8; j++) {
                const int m = m0 + 16 * mi + j + rhi;
                const int bidx = m >> 11, s = m & (SEQ - 1);
                const float c = acc[mi][ni][j] + bval;
                if (MODE == 0)
                    ((bf16*)dstv)[(((size_t)(bidx * N_HEADS + h) * SEQ + s) << 6) + d] =
                        (bf16)(c * 0.125f);
                else if (MODE == 1)
                    ((bf16*)dstv)[(((size_t)(bidx * N_HEADS + h) * SEQ + s) << 6) + d] = (bf16)c;
                else if (MODE == 2)
                    ((bf16*)dstv)[((size_t)(bidx * N_HEADS + h) * D_K + d) * SEQ + s] = (bf16)c;
                else
                    ((float*)dstv)[(size_t)m * D_MODEL + n] = c;
            }
        }
    }
}

// ---------------------------------------------------------------------------
// Flash attention: 128 threads = 4 waves, wave owns 16 queries; iterate over
// 32-key blocks with online softmax. P relayout (C-layout -> A-layout) via LDS.
// ---------------------------------------------------------------------------
__global__ __launch_bounds__(128) void attn_kernel(const bf16* __restrict__ qb,
                                                   const bf16* __restrict__ kb,
                                                   const bf16* __restrict__ vt,
                                                   bf16* __restrict__ attn_out) {
    __shared__ __align__(16) bf16 plds[4][16][40];   // padded stride, 16B-aligned rows

    const int wave = threadIdx.x >> 5;
    const int l = threadIdx.x & 31;
    const int bh = blockIdx.x >> 5;          // 64 (b,h) pairs
    const int qblk = blockIdx.x & 31;        // 32 query blocks of 64
    const int bidx = bh >> 4, h = bh & 15;
    const int q0 = qblk * 64 + wave * 16;

    const bf16* qbase = qb + ((size_t)bh * SEQ + q0) * D_K;
    const bf16* kbase = kb + (size_t)bh * SEQ * D_K;
    const bf16* vbase = vt + (size_t)bh * D_K * SEQ;

    // Q held in registers for the whole K sweep (scale already folded in)
    const v16bf qa0 = load_tileA(qbase, D_K);
    const v16bf qa1 = load_tileA(qbase + 32, D_K);

    v8f o[4];
    float mrow[8], lrow[8];
#pragma unroll
    for (int dt = 0; dt < 4; dt++)
#pragma unroll
        for (int j = 0; j < 8; j++) o[dt][j] = 0.f;
#pragma unroll
    for (int j = 0; j < 8; j++) { mrow[j] = -3.0e38f; lrow[j] = 0.f; }

    const float L2E = 1.44269504f;
    const int rhalf = (l >> 4) << 3;

    for (int kpos = 0; kpos < SEQ; kpos += 32) {
        v8f s0, s1;
#pragma unroll
        for (int j = 0; j < 8; j++) { s0[j] = 0.f; s1[j] = 0.f; }

        // scores = Q @ K^T  (K row-major [S,Dk] is exactly the B-operand of K^T)
        v16bf k00 = load_tileBT(kbase + (size_t)kpos * D_K, D_K);
        v16bf k01 = load_tileBT(kbase + (size_t)kpos * D_K + 32, D_K);
        v16bf k10 = load_tileBT(kbase + (size_t)(kpos + 16) * D_K, D_K);
        v16bf k11 = load_tileBT(kbase + (size_t)(kpos + 16) * D_K + 32, D_K);
        s0 = wmma_bf16(qa0, k00, s0);
        s0 = wmma_bf16(qa1, k01, s0);
        s1 = wmma_bf16(qa0, k10, s1);
        s1 = wmma_bf16(qa1, k11, s1);

        // online softmax: row stats live per-C-vgpr (row = j + 8*(l>>4)),
        // reduce across the 16 columns with half-wave butterflies
        float rmax[8], mnew[8], corr[8], psum[8];
#pragma unroll
        for (int j = 0; j < 8; j++) rmax[j] = fmaxf(s0[j], s1[j]);
#pragma unroll
        for (int sh = 1; sh <= 8; sh <<= 1)
#pragma unroll
            for (int j = 0; j < 8; j++)
                rmax[j] = fmaxf(rmax[j], __shfl_xor(rmax[j], sh, 32));
#pragma unroll
        for (int j = 0; j < 8; j++) {
            mnew[j] = fmaxf(mrow[j], rmax[j]);
            corr[j] = exp2f((mrow[j] - mnew[j]) * L2E);
        }
#pragma unroll
        for (int j = 0; j < 8; j++) {
            float p0 = exp2f((s0[j] - mnew[j]) * L2E);
            float p1 = exp2f((s1[j] - mnew[j]) * L2E);
            psum[j] = p0 + p1;
            plds[wave][j + rhalf][l & 15] = (bf16)p0;
            plds[wave][j + rhalf][16 + (l & 15)] = (bf16)p1;
        }
#pragma unroll
        for (int sh = 1; sh <= 8; sh <<= 1)
#pragma unroll
            for (int j = 0; j < 8; j++)
                psum[j] += __shfl_xor(psum[j], sh, 32);
#pragma unroll
        for (int j = 0; j < 8; j++) {
            lrow[j] = lrow[j] * corr[j] + psum[j];
            mrow[j] = mnew[j];
        }
#pragma unroll
        for (int dt = 0; dt < 4; dt++)
#pragma unroll
            for (int j = 0; j < 8; j++) o[dt][j] *= corr[j];

        __syncthreads();   // make P visible across lanes (cross-lane LDS RAW)

        // re-read P in A-operand layout (16 queries x 32 keys)
        v16bf pa;
        {
            ABReg t;
            const bf16* pp = &plds[wave][l & 15][(l >> 4) << 4];
            t.q128[0] = *(const uint4*)(pp);
            t.q128[1] = *(const uint4*)(pp + 8);
            pa = t.v;
        }
        // O += P @ V  (V stored transposed [Dk,S] -> contiguous B-operand reads)
#pragma unroll
        for (int dt = 0; dt < 4; dt++) {
            v16bf bv = load_tileBT(vbase + (size_t)(dt * 16) * SEQ + kpos, SEQ);
            o[dt] = wmma_bf16(pa, bv, o[dt]);
        }
        __syncthreads();
    }

    // epilogue: divide by row sums, store as [B,S,H*Dk] bf16 for out-proj GEMM
#pragma unroll
    for (int j = 0; j < 8; j++) {
        const float inv = 1.0f / lrow[j];
        const int m = q0 + j + rhalf;
#pragma unroll
        for (int dt = 0; dt < 4; dt++) {
            const int d = dt * 16 + (l & 15);
            attn_out[((size_t)(bidx * SEQ + m) * N_HEADS + h) * D_K + d] =
                (bf16)(o[dt][j] * inv);
        }
    }
}

// ---------------------------------------------------------------------------
extern "C" void kernel_launch(void* const* d_in, const int* in_sizes, int n_in,
                              void* d_out, int out_size, void* d_ws, size_t ws_size,
                              hipStream_t stream) {
    (void)in_sizes; (void)n_in; (void)out_size; (void)ws_size;
    const float* x  = (const float*)d_in[0];
    const float* wq = (const float*)d_in[1];
    const float* bq = (const float*)d_in[2];
    const float* wk = (const float*)d_in[3];
    const float* bk = (const float*)d_in[4];
    const float* wv = (const float*)d_in[5];
    const float* bv = (const float*)d_in[6];
    const float* wo = (const float*)d_in[7];
    const float* bo = (const float*)d_in[8];
    float* out = (float*)d_out;

    char* ws = (char*)d_ws;
    size_t off = 0;
    auto take = [&](size_t bytes) {
        char* p = ws + off;
        off += (bytes + 255) & ~(size_t)255;
        return (void*)p;
    };
    const size_t xbytes = (size_t)M_TOTAL * D_MODEL * sizeof(bf16);   // 16 MB
    const size_t wbytes = (size_t)D_MODEL * D_MODEL * sizeof(bf16);   //  2 MB
    bf16* xb    = (bf16*)take(xbytes);
    bf16* wqt   = (bf16*)take(wbytes);
    bf16* wkt   = (bf16*)take(wbytes);
    bf16* wvt   = (bf16*)take(wbytes);
    bf16* wot   = (bf16*)take(wbytes);
    bf16* qbuf  = (bf16*)take(xbytes);
    bf16* kbuf  = (bf16*)take(xbytes);
    bf16* vbuf  = (bf16*)take(xbytes);
    bf16* attnb = (bf16*)take(xbytes);

    // 1) convert inputs to bf16 (weights transposed for B-operand loads)
    cvt_x_kernel<<<(M_TOTAL * D_MODEL) / 256, 256, 0, stream>>>(x, xb);
    cvt_wT_kernel<<<(D_MODEL * D_MODEL) / 256, 256, 0, stream>>>(wq, wqt);
    cvt_wT_kernel<<<(D_MODEL * D_MODEL) / 256, 256, 0, stream>>>(wk, wkt);
    cvt_wT_kernel<<<(D_MODEL * D_MODEL) / 256, 256, 0, stream>>>(wv, wvt);
    cvt_wT_kernel<<<(D_MODEL * D_MODEL) / 256, 256, 0, stream>>>(wo, wot);

    // 2) QKV projections (bias fused; scale folded into Q; layouts permuted)
    dim3 gg(M_TOTAL / 256, D_MODEL / 64);
    gemm_kernel<0><<<gg, 256, 0, stream>>>(xb, wqt, bq, qbuf);
    gemm_kernel<1><<<gg, 256, 0, stream>>>(xb, wkt, bk, kbuf);
    gemm_kernel<2><<<gg, 256, 0, stream>>>(xb, wvt, bv, vbuf);

    // 3) flash attention
    attn_kernel<<<BATCH * N_HEADS * (SEQ / 64), 128, 0, stream>>>(qbuf, kbuf, vbuf, attnb);

    // 4) output projection -> f32 result
    gemm_kernel<3><<<gg, 256, 0, stream>>>(attnb, wot, bo, out);
}
